// NGPRadianceField_70471823393529
// MI455X (gfx1250) — matrix operand order repaired and checked by reading, not compiled, not run on hardware
//
#include <hip/hip_runtime.h>
#include <math.h>

typedef _Float16 f16;
typedef __attribute__((ext_vector_type(8)))  f16   v8h;
typedef __attribute__((ext_vector_type(16))) f16   v16h;
typedef __attribute__((ext_vector_type(8)))  float v8f;

#define T_SIZE   (1u << 19)
#define T_MASK   (T_SIZE - 1u)
#define PRIME1   2654435761u
#define PRIME2   805459861u

// Multi-resolution levels: floor(16 * (256^(1/15))^l)
__constant__ int c_res[16] = {16,23,33,48,70,101,147,212,307,445,645,933,1351,1955,2829,4096};

// ---- LDS layout (offsets in halfs) ----
#define W0T_OFF 0                 // 64 rows x 32  (w0:  K=32, N=64)
#define W1T_OFF 2048              // 64 rows x 64  (w1:  K=64, N=64)
#define WOT_OFF 6144              // 80 rows x 64  (w_out: K=64, N=65 -> pad 80)
#define H0T_OFF 11264             // 64 rows x 96  (h0:  K=80 -> pad 96, N=64)
#define H1T_OFF 17408             // 64 rows x 64  (h1:  K=64, N=64)
#define H2T_OFF 21504             // 16 rows x 64  (h2:  K=64, N=3 -> pad 16)
#define BUF_OFF 22528             // per-wave: 2 buffers of 32x96 halfs
#define NWAVES  4
#define BUF_HALFS (32*96)
#define DENS_OFF_BYTES ((BUF_OFF + NWAVES*2*BUF_HALFS)*2)
#define SMEM_BYTES (DENS_OFF_BYTES + NWAVES*32*(int)sizeof(float))

// Stage global f32 weight W[K][N] into LDS as transposed f16 Wt[nrows][kpad], zero padded.
__device__ __forceinline__ void stage_w(const float* __restrict__ W, f16* Wt,
                                        int K, int N, int kpad, int nrows, int tid) {
    for (int i = tid; i < nrows * kpad; i += 128) {
        int n = i / kpad;
        int k = i - n * kpad;
        float v = (n < N && k < K) ? W[k * N + n] : 0.0f;
        Wt[i] = (f16)v;
    }
}

// One 16x16 output tile of  A[32 x K] @ Wt^T, K = 32*KSTEPS.
// A: LDS activations, row stride astr (halfs).  Wt: LDS transposed weights, row stride kpad.
template<int KSTEPS>
__device__ __forceinline__ v8f wmma_tile(const f16* A, int astr, const f16* Wt, int kpad,
                                         int mtile, int n0, int lane) {
    v8f c = {};
    const int row  = (lane & 15) + mtile * 16;   // A row (point) this lane feeds
    const int asel = (lane >> 4) * 8;            // A: lanes 0-15 hold K=0..7&16..23; 16-31 hold 8..15&24..31
    const int ncol = n0 + (lane & 15);           // B column
    const int bsel = (lane >> 4) * 8;            // B: lanes 0-15 hold K=0..7&16..23; 16-31 hold 8..15&24..31
#pragma unroll
    for (int ks = 0; ks < KSTEPS; ++ks) {
        const int ko = ks * 32;
        const f16* ap = A + row * astr + ko + asel;
        v8h alo = *(const v8h*)(ap);
        v8h ahi = *(const v8h*)(ap + 16);
        const f16* bp = Wt + ncol * kpad + ko + bsel;
        v8h blo = *(const v8h*)(bp);
        v8h bhi = *(const v8h*)(bp + 16);
        v16h a, b;
#pragma unroll
        for (int i = 0; i < 8; ++i) { a[i] = alo[i]; a[i+8] = ahi[i]; b[i] = blo[i]; b[i+8] = bhi[i]; }
        c = __builtin_amdgcn_wmma_f32_16x16x32_f16(false, a, false, b, (short)0, c, false, false);
    }
    return c;
}

// Full layer: 32 points x Kin -> 32 points x (16*NTILES), optional ReLU, output to LDS f16.
template<int KSTEPS, int NTILES, bool RELU>
__device__ __forceinline__ void mlp_layer(const f16* A, int astr, const f16* Wt, int kpad,
                                          f16* O, int ostr, int lane) {
#pragma unroll
    for (int nt = 0; nt < NTILES; ++nt) {
#pragma unroll
        for (int mt = 0; mt < 2; ++mt) {
            v8f c = wmma_tile<KSTEPS>(A, astr, Wt, kpad, mt, nt * 16, lane);
            const int n  = nt * 16 + (lane & 15);
            const int mb = mt * 16 + ((lane >> 4) << 3);
#pragma unroll
            for (int r = 0; r < 8; ++r) {
                float v = c[r];
                if (RELU) v = v > 0.0f ? v : 0.0f;
                O[(mb + r) * ostr + n] = (f16)v;
            }
        }
    }
}

__global__ void __launch_bounds__(128) ngp_kernel(
    const float* __restrict__ positions, const float* __restrict__ directions,
    const float* __restrict__ table,
    const float* __restrict__ w0, const float* __restrict__ w1, const float* __restrict__ w_out,
    const float* __restrict__ h0, const float* __restrict__ h1, const float* __restrict__ h2,
    const float* __restrict__ aabb,
    float* __restrict__ out, int npts)
{
    extern __shared__ __align__(16) f16 smem[];
    const int tid  = threadIdx.x;
    const int lane = tid & 31;
    const int wave = tid >> 5;

    f16* w0t = smem + W0T_OFF;
    f16* w1t = smem + W1T_OFF;
    f16* wot = smem + WOT_OFF;
    f16* h0t = smem + H0T_OFF;
    f16* h1t = smem + H1T_OFF;
    f16* h2t = smem + H2T_OFF;
    f16* buf0 = smem + BUF_OFF + wave * 2 * BUF_HALFS;
    f16* buf1 = buf0 + BUF_HALFS;
    float* dens = (float*)((char*)smem + DENS_OFF_BYTES) + wave * 32;

    // ---- stage all weights (f32 -> f16 transposed) ----
    stage_w(w0,    w0t, 32, 64, 32, 64, tid);
    stage_w(w1,    w1t, 64, 64, 64, 64, tid);
    stage_w(w_out, wot, 64, 65, 64, 80, tid);
    stage_w(h0,    h0t, 80, 64, 96, 64, tid);
    stage_w(h1,    h1t, 64, 64, 64, 64, tid);
    stage_w(h2,    h2t, 64,  3, 64, 16, tid);

    // ---- per-lane point: hash-grid encode (scalar, L2-resident gathers) ----
    const int pt    = blockIdx.x * 128 + tid;
    const bool valid = pt < npts;
    const int ptc   = valid ? pt : (npts > 0 ? npts - 1 : 0);

    const float ax = aabb[0], ay = aabb[1], az = aabb[2];
    const float sx = 1.0f / (aabb[3] - ax);
    const float sy = 1.0f / (aabb[4] - ay);
    const float sz = 1.0f / (aabb[5] - az);
    const float px = (positions[ptc*3+0] - ax) * sx;
    const float py = (positions[ptc*3+1] - ay) * sy;
    const float pz = (positions[ptc*3+2] - az) * sz;

    for (int l = 0; l < 16; ++l) {
        const int res = c_res[l];
        const float fx = px * (float)res, fy = py * (float)res, fz = pz * (float)res;
        const float ffx = floorf(fx), ffy = floorf(fy), ffz = floorf(fz);
        const float wx = fx - ffx, wy = fy - ffy, wz = fz - ffz;
        const unsigned ix = (unsigned)(int)ffx, iy = (unsigned)(int)ffy, iz = (unsigned)(int)ffz;
        const bool dense = (l < 5);                 // (res+1)^3 <= T for levels 0..4
        const unsigned r1 = (unsigned)res + 1u;
        const float2* tb = (const float2*)(table + (size_t)l * T_SIZE * 2);
        float f0 = 0.0f, f1 = 0.0f;
#pragma unroll
        for (int c8 = 0; c8 < 8; ++c8) {
            const unsigned oi = c8 & 1, oj = (c8 >> 1) & 1, ok = (c8 >> 2) & 1;
            const unsigned cx = ix + oi, cy = iy + oj, cz = iz + ok;
            unsigned idx;
            if (dense) idx = cx + cy * r1 + cz * r1 * r1;
            else       idx = cx ^ (cy * PRIME1) ^ (cz * PRIME2);
            idx &= T_MASK;
            const float cw = (oi ? wx : 1.0f - wx) * (oj ? wy : 1.0f - wy) * (ok ? wz : 1.0f - wz);
            const float2 t = tb[idx];
            f0 += cw * t.x; f1 += cw * t.y;
        }
        buf0[lane * 96 + 2*l]     = (f16)f0;
        buf0[lane * 96 + 2*l + 1] = (f16)f1;
    }

    // ---- per-lane SH deg4 on normalized direction (kept in regs until buf1 is free) ----
    float sh[16];
    {
        const float dx = directions[ptc*3+0], dy = directions[ptc*3+1], dz = directions[ptc*3+2];
        const float inv = rsqrtf(dx*dx + dy*dy + dz*dz);
        const float x = dx*inv, y = dy*inv, z = dz*inv;
        const float xx = x*x, yy = y*y, zz = z*z;
        sh[0]  = 0.28209479177387814f;
        sh[1]  = -0.48860251190291987f * y;
        sh[2]  =  0.48860251190291987f * z;
        sh[3]  = -0.48860251190291987f * x;
        sh[4]  =  1.0925484305920792f * x * y;
        sh[5]  = -1.0925484305920792f * y * z;
        sh[6]  =  0.94617469575756f * zz - 0.31539156525252005f;
        sh[7]  = -1.0925484305920792f * x * z;
        sh[8]  =  0.5462742152960396f * (xx - yy);
        sh[9]  =  0.5900435899266435f * y * (3.0f * xx - yy);
        sh[10] =  2.890611442640554f * x * y * z;
        sh[11] =  0.4570457994644658f * y * (4.0f * zz - xx - yy);
        sh[12] =  0.3731763325901154f * z * (2.0f * zz - 3.0f * xx - 3.0f * yy);
        sh[13] =  0.4570457994644658f * x * (4.0f * zz - xx - yy);
        sh[14] =  1.445305721320277f * z * (xx - yy);
        sh[15] =  0.5900435899266435f * x * (xx - 3.0f * yy);
    }

    __syncthreads();  // weights + buf0 feats ready

    // layer0: feats[32] @ w0 -> relu -> buf1[.][0..63]
    mlp_layer<1, 4, true>(buf0, 96, w0t, 32, buf1, 96, lane);
    __syncthreads();
    // layer1: buf1 @ w1 -> relu -> buf0[.][0..63]
    mlp_layer<2, 4, true>(buf1, 96, w1t, 64, buf0, 96, lane);
    __syncthreads();

    // buf1 now free: seed it with [ SH(0..15) | geo(16..79) | zero pad(80..95) ]
#pragma unroll
    for (int i = 0; i < 16; ++i) buf1[lane * 96 + i] = (f16)sh[i];
#pragma unroll
    for (int i = 80; i < 96; ++i) buf1[lane * 96 + i] = (f16)0.0f;
    __syncthreads();

    // layer2 (w_out, 65 outputs): col 0 -> density (f32 LDS), cols 1..64 -> buf1[.][16..79]
#pragma unroll
    for (int nt = 0; nt < 5; ++nt) {
#pragma unroll
        for (int mt = 0; mt < 2; ++mt) {
            v8f c = wmma_tile<2>(buf0, 96, wot, 64, mt, nt * 16, lane);
            const int n  = nt * 16 + (lane & 15);
            const int mb = mt * 16 + ((lane >> 4) << 3);
#pragma unroll
            for (int r = 0; r < 8; ++r) {
                const int m = mb + r;
                if (n == 0)        dens[m] = c[r];
                else if (n <= 64)  buf1[m * 96 + 15 + n] = (f16)c[r];
            }
        }
    }
    __syncthreads();

    // color MLP: h0 (K=80 padded to 96) -> relu -> buf0
    mlp_layer<3, 4, true>(buf1, 96, h0t, 96, buf0, 96, lane);
    __syncthreads();
    // h1 -> relu -> buf1
    mlp_layer<2, 4, true>(buf0, 96, h1t, 64, buf1, 96, lane);
    __syncthreads();

    // h2: 3 outputs, sigmoid, straight to global
    const int wbase = blockIdx.x * 128 + wave * 32;
#pragma unroll
    for (int mt = 0; mt < 2; ++mt) {
        v8f c = wmma_tile<2>(buf1, 96, h2t, 64, mt, 0, lane);
        const int n  = lane & 15;
        const int mb = mt * 16 + ((lane >> 4) << 3);
        if (n < 3) {
#pragma unroll
            for (int r = 0; r < 8; ++r) {
                const int p = wbase + mb + r;
                if (p < npts) out[(size_t)p * 4 + n] = 1.0f / (1.0f + __expf(-c[r]));
            }
        }
    }
    // density: lane owns its own point/row
    if (valid) out[(size_t)pt * 4 + 3] = __expf(dens[lane] - 1.0f);
}

extern "C" void kernel_launch(void* const* d_in, const int* in_sizes, int n_in,
                              void* d_out, int out_size, void* d_ws, size_t ws_size,
                              hipStream_t stream) {
    (void)n_in; (void)out_size; (void)d_ws; (void)ws_size;
    const float* positions  = (const float*)d_in[0];
    const float* directions = (const float*)d_in[1];
    const float* table      = (const float*)d_in[2];
    const float* w0    = (const float*)d_in[3];
    const float* w1    = (const float*)d_in[4];
    const float* w_out = (const float*)d_in[5];
    const float* h0    = (const float*)d_in[6];
    const float* h1    = (const float*)d_in[7];
    const float* h2    = (const float*)d_in[8];
    const float* aabb  = (const float*)d_in[9];
    float* out = (float*)d_out;

    const int npts = in_sizes[0] / 3;
    const int blocks = (npts + 127) / 128;

    // >64KB dynamic LDS: raise the cap (host-side call, graph-capture safe, idempotent)
    hipFuncSetAttribute((const void*)ngp_kernel,
                        hipFuncAttributeMaxDynamicSharedMemorySize, SMEM_BYTES);

    ngp_kernel<<<blocks, 128, SMEM_BYTES, stream>>>(
        positions, directions, table, w0, w1, w_out, h0, h1, h2, aabb, out, npts);
}
